// CrossLevelAttention_20151986553270
// MI455X (gfx1250) — matrix operand order repaired and checked by reading, not compiled, not run on hardware
//
#include <hip/hip_runtime.h>
#include <hip/hip_bf16.h>

typedef unsigned short u16;
typedef unsigned int   u32;
typedef __attribute__((ext_vector_type(16))) __bf16 v16bf;
typedef __attribute__((ext_vector_type(8)))  float  v8f;

#define NC 200000
#define NT 2000
#define DH 256

// ---------------- helpers ----------------
__device__ __forceinline__ u16 f2bf(float f) {          // f32 -> bf16 RNE
  u32 u = __float_as_uint(f);
  return (u16)((u + 0x7FFFu + ((u >> 16) & 1u)) >> 16);
}

union Frag { v16bf v; uint4 q[2]; };

// Async Global->LDS 128-bit DMA copy (CDNA5, tracked by ASYNCcnt).
// lds_off = LDS byte offset (low 32 bits of flat shared address).
__device__ __forceinline__ void async_copy_b128(unsigned lds_off, const void* gptr) {
  asm volatile("global_load_async_to_lds_b128 %0, %1, off"
               :: "v"(lds_off), "v"(gptr)
               : "memory");
}
__device__ __forceinline__ void wait_asynccnt0() {
  asm volatile("s_wait_asynccnt 0x0" ::: "memory");
}

// ---------------- tiny kernels ----------------
__global__ void k_zero(float* __restrict__ p, int n) {
  int i = blockIdx.x * 256 + threadIdx.x;
  if (i < n) p[i] = 0.f;
}

// W [K=256][N=256] f32 -> Wt [N][K] bf16 (contiguous K per output column)
__global__ void k_convert_w(const float* __restrict__ W, u16* __restrict__ Wt) {
  int n = blockIdx.x, k = threadIdx.x;
  Wt[n * 256 + k] = f2bf(W[k * 256 + n]);
}

// small GEMM: Y[t][h] = sum_k X[t][k]*W[k][h] + b[h]   (T=2000, trivial FLOPs)
__global__ __launch_bounds__(256) void k_small_gemm(const float* __restrict__ X,
                                                    const float* __restrict__ W,
                                                    const float* __restrict__ b,
                                                    float* __restrict__ Y) {
  __shared__ float xr[256];
  int t = blockIdx.x, h = threadIdx.x;
  xr[h] = X[t * 256 + h];
  __syncthreads();
  float s = b[h];
#pragma unroll 8
  for (int k = 0; k < 256; ++k) s = fmaf(xr[k], W[k * 256 + h], s);
  Y[t * 256 + h] = s;
}

__global__ void k_segmax(const float* __restrict__ sc, const int* __restrict__ seg,
                         u32* __restrict__ mb, int n) {
  int i = blockIdx.x * 256 + threadIdx.x;
  if (i >= n) return;
  u32 u = __float_as_uint(sc[i]);
  u = (u & 0x80000000u) ? ~u : (u | 0x80000000u);   // order-preserving encode
  atomicMax(&mb[seg[i]], u);
}

__global__ void k_exp(float* __restrict__ sc, const int* __restrict__ seg,
                      const u32* __restrict__ mb, float* __restrict__ den, int n) {
  int i = blockIdx.x * 256 + threadIdx.x;
  if (i >= n) return;
  int s = seg[i];
  u32 enc = mb[s];
  float m = __uint_as_float((enc & 0x80000000u) ? (enc & 0x7FFFFFFFu) : ~enc);
  float e = __expf(sc[i] - m);
  sc[i] = e;
  unsafeAtomicAdd(&den[s], e);
}

// one wave per cell: agg[seg[c]][:] += cf[c][:] * (e[c]/den[seg[c]])
__global__ __launch_bounds__(256) void k_agg(const float* __restrict__ cf,
                                             const float* __restrict__ e,
                                             const float* __restrict__ den,
                                             const int* __restrict__ seg,
                                             float* __restrict__ agg, int n) {
  int wv = threadIdx.x >> 5, lane = threadIdx.x & 31;
  int c = blockIdx.x * 8 + wv;
  if (c >= n) return;
  int s = seg[c];
  float w = e[c] / den[s];
  const float4* src = (const float4*)&cf[c * 256 + lane * 8];
  float4 x0 = src[0], x1 = src[1];
  float* dst = &agg[s * 256 + lane * 8];
  unsafeAtomicAdd(dst + 0, x0.x * w); unsafeAtomicAdd(dst + 1, x0.y * w);
  unsafeAtomicAdd(dst + 2, x0.z * w); unsafeAtomicAdd(dst + 3, x0.w * w);
  unsafeAtomicAdd(dst + 4, x1.x * w); unsafeAtomicAdd(dst + 5, x1.y * w);
  unsafeAtomicAdd(dst + 6, x1.z * w); unsafeAtomicAdd(dst + 7, x1.w * w);
}

__global__ void k_tissue_upd(const float* __restrict__ tf, const float* __restrict__ agg,
                             float* __restrict__ tu, float* __restrict__ out_t, int n) {
  int i = blockIdx.x * 256 + threadIdx.x;
  if (i < n) { float v = tf[i] + agg[i]; tu[i] = v; out_t[i] = v; }
}

// ---------------- fused WMMA GEMM + attention epilogue ----------------
// Block: 256 thr = 8 waves (2 M-waves x 4 N-waves), tile M=32 x H=256, K=256.
// mode 0: out = scores[c] = tanh(cf@Wt + bias + trow[seg]) . avec
// mode 1: gate = sigmoid(score); out[c][:] = cf[c][:] + tu[seg[c]][:] * gate
__global__ __launch_bounds__(256) void k_fused(const float* __restrict__ cf,
                                               const int*   __restrict__ seg,
                                               const u16*   __restrict__ Wt,   // [N][K] bf16
                                               const float* __restrict__ bias,
                                               const float* __restrict__ avec,
                                               const float* __restrict__ trow, // [NT][256]
                                               const float* __restrict__ tu,   // [NT][256] (mode1)
                                               float* __restrict__ outv,
                                               int mode) {
  __shared__ u16   As[32 * 32];        // bf16 A tile, 32 rows x 32 k
  __shared__ float tqs[32][256];       // gathered trow[seg] rows (async DMA)
  __shared__ float bLDS[256], aLDS[256];
  __shared__ int   segLDS[32];
  __shared__ float scoreP[32][4];
  __shared__ float gateLDS[32];

  const int t = threadIdx.x;
  const int m0 = blockIdx.x * 32;
  const int wv = t >> 5, lane = t & 31;
  const int wm = wv >> 2, wn = wv & 3;     // 2 x 4 wave grid
  const int half = lane >> 4, lr = lane & 15;

  if (t < 32) segLDS[t] = seg[m0 + t];
  bLDS[t] = bias[t];
  aLDS[t] = avec[t];
  __syncthreads();

  // kick off the 32 gathered trow rows as async Global->LDS DMA copies;
  // they complete in the shadow of the whole GEMM K-loop below.
#pragma unroll
  for (int i = 0; i < 8; ++i) {
    int e = i * 1024 + t * 4;
    int row = e >> 8, col = e & 255;
    unsigned lds_off = (unsigned)(unsigned long long)&tqs[row][col];
    async_copy_b128(lds_off, &trow[segLDS[row] * 256 + col]);
  }

  v8f acc[4] = {};
  const float* arow = cf + m0 * 256;

  for (int kt = 0; kt < 8; ++kt) {
    const int k0 = kt * 32;
    __syncthreads();
    { // stage A: 1024 bf16, 4 per thread
      int idx = t * 4;
      int row = idx >> 5, kk = idx & 31;
      float4 f = *(const float4*)&arow[row * 256 + k0 + kk];
      ushort4 h;
      h.x = f2bf(f.x); h.y = f2bf(f.y); h.z = f2bf(f.z); h.w = f2bf(f.w);
      *(ushort4*)&As[row * 32 + kk] = h;
    }
    __syncthreads();

    // A fragment per ISA layout: lanes 0-15 K{0-7,16-23}, lanes 16-31 K{8-15,24-31}
    Frag a;
    {
      int row = wm * 16 + lr;
      a.q[0] = *(const uint4*)&As[row * 32 + half * 8];
      a.q[1] = *(const uint4*)&As[row * 32 + 16 + half * 8];
    }
#pragma unroll
    for (int nt = 0; nt < 4; ++nt) {
      // B fragment: n = lane&15 (+tile), lanes 0-15 K{0-7,16-23}, lanes 16-31 K{8-15,24-31}
      int n = wn * 64 + nt * 16 + lr;
      Frag b;
      const u16* bp = &Wt[n * 256 + k0 + half * 8];
      b.q[0] = *(const uint4*)&bp[0];
      b.q[1] = *(const uint4*)&bp[16];
      acc[nt] = __builtin_amdgcn_wmma_f32_16x16x32_bf16(
          false, a.v, false, b.v, (short)0, acc[nt], false, false);
    }
  }

  // make the async tqs gather visible to the whole block
  wait_asynccnt0();
  __syncthreads();

  // epilogue: D layout m = v + 8*half, n = lr (+tile offsets)
  float part[8];
#pragma unroll
  for (int v = 0; v < 8; ++v) {
    int mrow = wm * 16 + v + half * 8;
    float p = 0.f;
#pragma unroll
    for (int nt = 0; nt < 4; ++nt) {
      int n = wn * 64 + nt * 16 + lr;
      float val = acc[nt][v] + bLDS[n] + tqs[mrow][n];
      p += tanhf(val) * aLDS[n];
    }
    // reduce across the 16 lanes sharing this half (xor stays within group)
    p += __shfl_xor(p, 1);
    p += __shfl_xor(p, 2);
    p += __shfl_xor(p, 4);
    p += __shfl_xor(p, 8);
    part[v] = p;
  }
  if (lr == 0) {
#pragma unroll
    for (int v = 0; v < 8; ++v)
      scoreP[wm * 16 + v + half * 8][wn] = part[v];
  }
  __syncthreads();

  if (mode == 0) {
    if (t < 32)
      outv[m0 + t] = scoreP[t][0] + scoreP[t][1] + scoreP[t][2] + scoreP[t][3];
  } else {
    if (t < 32) {
      float s = scoreP[t][0] + scoreP[t][1] + scoreP[t][2] + scoreP[t][3];
      gateLDS[t] = 1.f / (1.f + __expf(-s));
    }
    __syncthreads();
#pragma unroll
    for (int i = 0; i < 8; ++i) {
      int e = i * 1024 + t * 4;
      int row = e >> 8, col = e & 255;
      float4 c = *(const float4*)&cf[(m0 + row) * 256 + col];
      float4 u = *(const float4*)&tu[segLDS[row] * 256 + col];
      float g = gateLDS[row];
      float4 o;
      o.x = c.x + u.x * g; o.y = c.y + u.y * g;
      o.z = c.z + u.z * g; o.w = c.w + u.w * g;
      *(float4*)&outv[(m0 + row) * 256 + col] = o;
    }
  }
}

// ---------------- launch ----------------
extern "C" void kernel_launch(void* const* d_in, const int* in_sizes, int n_in,
                              void* d_out, int out_size, void* d_ws, size_t ws_size,
                              hipStream_t stream) {
  const float* cf  = (const float*)d_in[0];
  const float* tf  = (const float*)d_in[1];
  const int*   seg = (const int*)d_in[2];
  const float* Wcp = (const float*)d_in[3];
  const float* bcp = (const float*)d_in[4];
  const float* Wtq = (const float*)d_in[5];
  const float* btq = (const float*)d_in[6];
  const float* abu = (const float*)d_in[7];
  const float* Wtp = (const float*)d_in[8];
  const float* btp = (const float*)d_in[9];
  const float* Wcq = (const float*)d_in[10];
  const float* bcq = (const float*)d_in[11];
  const float* atd = (const float*)d_in[12];

  char* ws = (char*)d_ws;
  u16*   wt_bu = (u16*)(ws + 0);          // 131072 B
  u16*   wt_td = (u16*)(ws + 131072);     // 131072 B
  float* tq    = (float*)(ws + 262144);   // 2048000 B
  float* tk    = (float*)(ws + 2310144);  // 2048000 B
  float* tu    = (float*)(ws + 4358144);  // 2048000 B
  float* sc    = (float*)(ws + 6406144);  //  800000 B
  float* agg   = (float*)(ws + 7206144);  // 2048000 B  -- zeroed region start
  float* den   = (float*)(ws + 9254144);  //    8000 B
  u32*   mb    = (u32*)(ws + 9262144);    //    8000 B

  float* out_cell = (float*)d_out;
  float* out_tis  = out_cell + (size_t)NC * DH;

  // zero agg | den | mb (contiguous 516000 words)
  k_zero<<<dim3((516000 + 255) / 256), dim3(256), 0, stream>>>(agg, 516000);
  // bf16 transposed weight copies for the two big GEMMs
  k_convert_w<<<dim3(256), dim3(256), 0, stream>>>(Wcp, wt_bu);
  k_convert_w<<<dim3(256), dim3(256), 0, stream>>>(Wcq, wt_td);
  // t_query = tissue @ W_tissue_query + b
  k_small_gemm<<<dim3(NT), dim3(256), 0, stream>>>(tf, Wtq, btq, tq);
  // bottom-up fused GEMM + scores
  k_fused<<<dim3(NC / 32), dim3(256), 0, stream>>>(cf, seg, wt_bu, bcp, abu, tq,
                                                   nullptr, sc, 0);
  // segment softmax
  k_segmax<<<dim3((NC + 255) / 256), dim3(256), 0, stream>>>(sc, seg, mb, NC);
  k_exp<<<dim3((NC + 255) / 256), dim3(256), 0, stream>>>(sc, seg, mb, den, NC);
  // weighted segment sum of cell features
  k_agg<<<dim3((NC + 7) / 8), dim3(256), 0, stream>>>(cf, sc, den, seg, agg, NC);
  // tissue_updated (also written to output slot)
  k_tissue_upd<<<dim3(NT * DH / 256), dim3(256), 0, stream>>>(tf, agg, tu, out_tis,
                                                              NT * DH);
  // t_key = tissue_updated @ W_tissue_proj + b
  k_small_gemm<<<dim3(NT), dim3(256), 0, stream>>>(tu, Wtp, btp, tk);
  // top-down fused GEMM + gate + final cell update
  k_fused<<<dim3(NC / 32), dim3(256), 0, stream>>>(cf, seg, wt_td, bcq, atd, tk,
                                                   tu, out_cell, 1);
}
